// ArcFace_27032524161589
// MI455X (gfx1250) — matrix operand order repaired and checked by reading, not compiled, not run on hardware
//
#include <hip/hip_runtime.h>
#include <hip/hip_bf16.h>
#include <math.h>

#define IN_DIM   512
#define OUT_DIM  100000
#define BATCH    512
#define S_SCALE  64.0f
#define MARGIN_M 0.5f
#define NC       10
#define EPS_N    1e-12f

typedef __attribute__((ext_vector_type(16))) __bf16 v16bf;
typedef __attribute__((ext_vector_type(8)))  float  v8f;

union VecBF { v16bf v; unsigned short u[16]; };

__device__ __forceinline__ unsigned short f32_to_bf16_bits(float f) {
    union { float f; unsigned int u; } a; a.f = f;
    unsigned int r = a.u + 0x7FFFu + ((a.u >> 16) & 1u);  // round-to-nearest-even
    return (unsigned short)(r >> 16);
}
__device__ __forceinline__ float bf16_bits_to_f32(unsigned short h) {
    union { unsigned int u; float f; } a; a.u = ((unsigned int)h) << 16;
    return a.f;
}

// A-pack layout: fragment index = (ks*32 + mt), 512 bf16 per fragment,
// within fragment: lane*16 + j  (contiguous k = 16*(lane/16) + j order, matched by B).
__device__ __forceinline__ int a_pack_index(int m, int k) {
    int mt = m >> 4, ks = k >> 5;
    int lane = (m & 15) | (((k >> 4) & 1) << 4);
    int j    = (k & 15);
    return ((ks * 32 + mt) * 512) + lane * 16 + j;
}

// ---------------- Kernel 1: row L2-norm of x + pack to bf16 hi/lo fragments --
__global__ void __launch_bounds__(256)
arcface_rownorm_pack(const float* __restrict__ x,
                     unsigned short* __restrict__ Ahi,
                     unsigned short* __restrict__ Alo) {
    __shared__ float red[256];
    __shared__ float sinv;
    const int m = blockIdx.x;
    const int t = threadIdx.x;
    float v0 = x[m * IN_DIM + t];
    float v1 = x[m * IN_DIM + t + 256];
    red[t] = v0 * v0 + v1 * v1;
    __syncthreads();
    for (int s = 128; s > 0; s >>= 1) {
        if (t < s) red[t] += red[t + s];
        __syncthreads();
    }
    if (t == 0) sinv = 1.0f / fmaxf(sqrtf(red[0]), EPS_N);
    __syncthreads();
    const float inv = sinv;

    float vals[2] = { v0 * inv, v1 * inv };
    int   ks_ [2] = { t, t + 256 };
    #pragma unroll
    for (int i = 0; i < 2; ++i) {
        int k = ks_[i];
        unsigned short hi = f32_to_bf16_bits(vals[i]);
        float hif = bf16_bits_to_f32(hi);
        unsigned short lo = f32_to_bf16_bits(vals[i] - hif);
        int idx = a_pack_index(m, k);
        Ahi[idx] = hi;
        Alo[idx] = lo;
    }
}

// ---------------- Kernel 2: inverse column L2-norms of W ---------------------
__global__ void __launch_bounds__(256)
arcface_colnorm(const float* __restrict__ W, float* __restrict__ invWn) {
    int c = blockIdx.x * 256 + threadIdx.x;
    if (c >= OUT_DIM) return;
    float ss = 0.0f;
    #pragma unroll 4
    for (int i = 0; i < IN_DIM; ++i) {
        float v = W[(size_t)i * OUT_DIM + c];
        ss += v * v;
    }
    invWn[c] = 1.0f / fmaxf(sqrtf(ss), EPS_N);
}

// ---------------- Kernel 3: split-bf16 WMMA cosine GEMM ----------------------
// Block tile: 128 (M) x 64 (N). 8 wave32s: wm = wave>>1 (4 x 32 rows),
// wn = wave&1 (2 x 32 cols). Each wave owns 2x2 16x16 accumulator tiles.
// W tiles are converted to bf16 hi/lo once during staging into DOUBLE-BUFFERED
// LDS; the WMMA phase gathers B fragments with pure ds_load_b128. Staging uses
// clamped addresses + selects (no exec-mask predication in the hot loop).
#define BK_STRIDE 40   // LDS row stride in bf16 elems ([n][k] layout, 32 k + pad)
#define NKSTEPS   (IN_DIM / 32)

__device__ __forceinline__ void stage_tile(const float* __restrict__ W,
                                           int kg0, int sn0, int sg0,
                                           bool gc_ok, size_t cbase,
                                           unsigned short* __restrict__ bhi,
                                           unsigned short* __restrict__ blo) {
    #pragma unroll
    for (int it = 0; it < 2; ++it) {
        const int g = sg0 + it * 4;             // k-group 0..7 (4 rows each)
        float v[4];
        #pragma unroll
        for (int j = 0; j < 4; ++j) {
            int kr = kg0 + g * 4 + j;
            float raw = W[(size_t)kr * OUT_DIM + cbase];   // always in-bounds
            v[j] = gc_ok ? raw : 0.0f;                     // v_cndmask, no branch
        }
        unsigned long long ph = 0ull, pl = 0ull;
        #pragma unroll
        for (int j = 0; j < 4; ++j) {
            unsigned short h = f32_to_bf16_bits(v[j]);
            unsigned short l = f32_to_bf16_bits(v[j] - bf16_bits_to_f32(h));
            ph |= ((unsigned long long)h) << (16 * j);
            pl |= ((unsigned long long)l) << (16 * j);
        }
        *(unsigned long long*)&bhi[sn0 * BK_STRIDE + g * 4] = ph;
        *(unsigned long long*)&blo[sn0 * BK_STRIDE + g * 4] = pl;
    }
}

__global__ void __launch_bounds__(256)
arcface_gemm_wmma(const unsigned short* __restrict__ Ahi,
                  const unsigned short* __restrict__ Alo,
                  const float* __restrict__ W,
                  const float* __restrict__ invWn,
                  float* __restrict__ out) {
    __shared__ unsigned short ldsBhi[2][64 * BK_STRIDE];   // 2 x 5120 B
    __shared__ unsigned short ldsBlo[2][64 * BK_STRIDE];   // 2 x 5120 B

    const int nb = blockIdx.x * 64;
    const int mb = blockIdx.y * 128;
    const int tid  = threadIdx.x;
    const int wave = tid >> 5;
    const int lane = tid & 31;
    const int wm = wave >> 1;
    const int wn = wave & 1;
    const int mt0 = (mb >> 4) + wm * 2;     // first 16-row tile index (global)

    const v8f vzero = { 0.f, 0.f, 0.f, 0.f, 0.f, 0.f, 0.f, 0.f };
    v8f acc[2][2];
    acc[0][0] = vzero; acc[0][1] = vzero; acc[1][0] = vzero; acc[1][1] = vzero;

    const int kb = (lane >> 4) * 16;        // K half for the B fragment

    // staging decomposition: 512 slots = 64 cols x 8 groups of 4 k-rows
    const int  sn0   = tid & 63;            // column within tile
    const int  sg0   = tid >> 6;            // base k-group 0..3
    const int  gc    = nb + sn0;
    const bool gc_ok = gc < OUT_DIM;
    const size_t cbase = (size_t)(gc_ok ? gc : (OUT_DIM - 1));

    // prologue: stage tile 0 into buffer 0
    stage_tile(W, 0, sn0, sg0, gc_ok, cbase, ldsBhi[0], ldsBlo[0]);

    for (int ks = 0; ks < NKSTEPS; ++ks) {
        __syncthreads();                    // staging of buffer ks&1 visible

        // issue staging for the NEXT tile first (global latency hides under WMMA)
        if (ks + 1 < NKSTEPS) {
            stage_tile(W, (ks + 1) * 32, sn0, sg0, gc_ok, cbase,
                       ldsBhi[(ks + 1) & 1], ldsBlo[(ks + 1) & 1]);
        }

        const unsigned short* bh = ldsBhi[ks & 1];
        const unsigned short* bl = ldsBlo[ks & 1];

        // ---- load pre-packed A fragments (hi/lo) ---------------------------
        VecBF ahi[2], alo[2];
        #pragma unroll
        for (int mi = 0; mi < 2; ++mi) {
            int off = ((ks * 32 + (mt0 + mi)) * 512) + lane * 16;
            ahi[mi].v = *(const v16bf*)(Ahi + off);
            alo[mi].v = *(const v16bf*)(Alo + off);
        }

        // ---- per 16-col tile: B fragments straight from LDS, then WMMA -----
        #pragma unroll
        for (int ni = 0; ni < 2; ++ni) {
            const int nloc = wn * 32 + ni * 16 + (lane & 15);
            VecBF bhi, blo;
            bhi.v = *(const v16bf*)&bh[nloc * BK_STRIDE + kb];
            blo.v = *(const v16bf*)&bl[nloc * BK_STRIDE + kb];
            #pragma unroll
            for (int mi = 0; mi < 2; ++mi) {
                acc[mi][ni] = __builtin_amdgcn_wmma_f32_16x16x32_bf16(
                    false, ahi[mi].v, false, bhi.v, (short)0, acc[mi][ni], false, false);
                acc[mi][ni] = __builtin_amdgcn_wmma_f32_16x16x32_bf16(
                    false, alo[mi].v, false, bhi.v, (short)0, acc[mi][ni], false, false);
                acc[mi][ni] = __builtin_amdgcn_wmma_f32_16x16x32_bf16(
                    false, ahi[mi].v, false, blo.v, (short)0, acc[mi][ni], false, false);
            }
        }
    }

    // ---- epilogue: apply column inv-norm, clamp, scale, store ---------------
    #pragma unroll
    for (int ni = 0; ni < 2; ++ni) {
        int col = nb + wn * 32 + ni * 16 + (lane & 15);
        if (col >= OUT_DIM) continue;
        float invw = invWn[col];
        #pragma unroll
        for (int mi = 0; mi < 2; ++mi) {
            int row0 = mb + wm * 32 + mi * 16 + (lane >> 4) * 8;
            #pragma unroll
            for (int r = 0; r < 8; ++r) {
                float c = fminf(1.0f, fmaxf(-1.0f, acc[mi][ni][r] * invw));
                out[(size_t)(row0 + r) * OUT_DIM + col] = S_SCALE * c;
            }
        }
    }
}

// ---------------- Kernel 4: angular margin + intra score + top-NC -----------
__global__ void __launch_bounds__(256)
arcface_margin_topk(float* __restrict__ logits,
                    const int* __restrict__ y,
                    float* __restrict__ intra,
                    float* __restrict__ hard) {
    const int row = blockIdx.x;
    const int t   = threadIdx.x;
    const int yc  = y[row];
    float* lrow = logits + (size_t)row * OUT_DIM;

    // thread 0: margin substitution + intra score (others never touch col yc)
    if (t == 0) {
        float cost  = fminf(1.0f, fmaxf(-1.0f, lrow[yc] / S_SCALE));
        float theta = acosf(cost);
        float mc    = (theta < ((float)M_PI - MARGIN_M))
                        ? cosf(theta + MARGIN_M)
                        : (cost - sinf(MARGIN_M) * MARGIN_M);
        lrow[yc]   = S_SCALE * mc;
        intra[row] = cost;
    }

    // per-thread sorted top-NC over strided columns (values are S*cosine;
    // ordering is preserved since S > 0), excluding the target column
    float tv[NC]; int ti[NC];
    #pragma unroll
    for (int i = 0; i < NC; ++i) { tv[i] = -1e30f; ti[i] = 0; }
    for (int c = t; c < OUT_DIM; c += 256) {
        if (c == yc) continue;
        float v = lrow[c];
        if (v > tv[NC - 1]) {
            tv[NC - 1] = v; ti[NC - 1] = c;
            #pragma unroll
            for (int p = NC - 1; p > 0; --p) {
                if (tv[p] > tv[p - 1]) {
                    float fv = tv[p]; tv[p] = tv[p - 1]; tv[p - 1] = fv;
                    int   fi = ti[p]; ti[p] = ti[p - 1]; ti[p - 1] = fi;
                }
            }
        }
    }

    // tournament merge: NC rounds of argmax over each thread's current head
    __shared__ float rv[256];
    __shared__ int   ri[256];
    __shared__ int   wcls;
    int p = 0;
    for (int round = 0; round < NC; ++round) {
        rv[t] = (p < NC) ? tv[p] : -1e30f;
        ri[t] = t;
        __syncthreads();
        for (int s = 128; s > 0; s >>= 1) {
            if (t < s) {
                if (rv[t + s] > rv[t]) { rv[t] = rv[t + s]; ri[t] = ri[t + s]; }
            }
            __syncthreads();
        }
        int wtid = ri[0];
        if (t == wtid) { wcls = ti[p]; ++p; }
        __syncthreads();
        if (t == 0) hard[row * NC + round] = (float)wcls;
        __syncthreads();
    }
}

// ---------------- launcher ---------------------------------------------------
extern "C" void kernel_launch(void* const* d_in, const int* in_sizes, int n_in,
                              void* d_out, int out_size, void* d_ws, size_t ws_size,
                              hipStream_t stream) {
    (void)in_sizes; (void)n_in; (void)out_size; (void)ws_size;
    const float* x = (const float*)d_in[0];
    const float* W = (const float*)d_in[1];
    const int*   y = (const int*)d_in[2];

    float* logits = (float*)d_out;                                   // [512,100000]
    float* intra  = logits + (size_t)BATCH * OUT_DIM;                // [512]
    float* hard   = intra + BATCH;                                   // [512,10]

    unsigned short* Ahi  = (unsigned short*)d_ws;                    // 512 KB
    unsigned short* Alo  = Ahi + (size_t)BATCH * IN_DIM;             // 512 KB
    float*          invW = (float*)((char*)d_ws + 2u * BATCH * IN_DIM * sizeof(unsigned short));

    arcface_rownorm_pack<<<BATCH, 256, 0, stream>>>(x, Ahi, Alo);
    arcface_colnorm<<<(OUT_DIM + 255) / 256, 256, 0, stream>>>(W, invW);
    dim3 gg((OUT_DIM + 63) / 64, BATCH / 128);
    arcface_gemm_wmma<<<gg, 256, 0, stream>>>(Ahi, Alo, W, invW, logits);
    arcface_margin_topk<<<BATCH, 256, 0, stream>>>(logits, y, intra, hard);
}
